// MultiHeadBiaffine_61400852463801
// MI455X (gfx1250) — matrix operand order repaired and checked by reading, not compiled, not run on hardware
//
#include <hip/hip_runtime.h>

typedef __attribute__((ext_vector_type(16))) _Float16 v16h;
typedef __attribute__((ext_vector_type(8)))  _Float16 h8;
typedef __attribute__((ext_vector_type(8)))  float    v8f;
typedef __attribute__((ext_vector_type(8)))  float    f8;
typedef __attribute__((ext_vector_type(4)))  int      i4;

#define AS_GLOBAL __attribute__((address_space(1)))
#define AS_LDS    __attribute__((address_space(3)))

#define WMMA_F16(a, b, c) \
  __builtin_amdgcn_wmma_f32_16x16x32_f16(false, (a), false, (b), (short)0, (c), false, false)

// ---- gfx1250 async global->LDS copy (guarded; falls back to plain copy) ----
#if defined(__has_builtin)
#if __has_builtin(__builtin_amdgcn_global_load_async_to_lds_b128)
#define HAVE_ASYNC_LDS 1
#endif
#endif
#ifndef HAVE_ASYNC_LDS
#define HAVE_ASYNC_LDS 0
#endif

#if defined(__has_builtin)
#if __has_builtin(__builtin_amdgcn_s_wait_asynccnt)
#define ASYNC_WAIT() __builtin_amdgcn_s_wait_asynccnt(0)
#endif
#endif
#ifndef ASYNC_WAIT
#define ASYNC_WAIT() asm volatile("s_wait_asynccnt 0x0" ::: "memory")
#endif

// ---- A-fragment (16-bit 16x32, MxK): lane = 16*g + m ; elems 0..15 -> K = kb + 16g + e (contiguous)
__device__ __forceinline__ v16h fragA_f16(const _Float16* p, int kb, int g) {
  h8 lo = *(const h8*)(p + kb + 16 * g);
  h8 hi = *(const h8*)(p + kb + 16 * g + 8);
  return __builtin_shufflevector(lo, hi, 0,1,2,3,4,5,6,7,8,9,10,11,12,13,14,15);
}
__device__ __forceinline__ v16h fragA_f32(const float* p, int kb, int g) {
  f8 lo = *(const f8*)(p + kb + 16 * g);
  f8 hi = *(const f8*)(p + kb + 16 * g + 8);
  v16h r;
#pragma unroll
  for (int i = 0; i < 8; ++i) { r[i] = (_Float16)lo[i]; r[i + 8] = (_Float16)hi[i]; }
  return r;
}
// ---- B-fragment (16-bit 32x16, KxN): lane = 16*g + n ; elems 0..15 -> K = kb + 16g + e (contiguous)
__device__ __forceinline__ v16h fragB_f16(const _Float16* p, int kb, int g) {
  return *(const v16h*)(p + kb + 16 * g);      // 32B-aligned (global layouts)
}
__device__ __forceinline__ v16h fragB_16al(const _Float16* p, int kb, int g) {
  h8 lo = *(const h8*)(p + kb + 16 * g);       // 16B-aligned variant (padded LDS rows)
  h8 hi = *(const h8*)(p + kb + 16 * g + 8);
  return __builtin_shufflevector(lo, hi, 0,1,2,3,4,5,6,7,8,9,10,11,12,13,14,15);
}

// ============ Kernel 1: convert Wh / Wv (512x1024 each) f32 -> f16, same layout ============
__global__ __launch_bounds__(256) void k_cvt_lin(const float* __restrict__ Wh,
                                                 const float* __restrict__ Wv,
                                                 _Float16* __restrict__ Whf,
                                                 _Float16* __restrict__ Wvf) {
  const int N = 512 * 1024;
  int i = blockIdx.x * 256 + threadIdx.x;
  if (i < N)       Whf[i] = (_Float16)Wh[i];
  else             Wvf[i - N] = (_Float16)Wv[i - N];
}

// ============ Kernel 2: Wt[h][d][y][x] = (f16) W[h][d][x][y]  (LDS tile transpose) ============
__global__ __launch_bounds__(256) void k_cvt_W(const float* __restrict__ W,
                                               _Float16* __restrict__ Wt) {
  __shared__ _Float16 t[128 * 130];              // padded rows: bank-conflict free
  const int hd = blockIdx.x;                     // h*128 + d, 512 blocks
  const float* in = W + (size_t)hd * 16384;      // [x][y], y fastest
  _Float16* out = Wt + (size_t)hd * 16384;       // [y][x], x fastest
  for (int i = threadIdx.x; i < 16384; i += 256) {
    int x = i >> 7, y = i & 127;
    t[y * 130 + x] = (_Float16)in[i];
  }
  __syncthreads();
  for (int i = threadIdx.x; i < 16384; i += 256) {
    int y = i >> 7, x = i & 127;
    out[i] = t[y * 130 + x];
  }
}

// ============ Kernel 3: h/v = leaky_relu(x @ W^T + b), f16 out ============
// grid (32, 2): x-> 32 row-blocks of 32; y-> 0:h, 1:v.  8 waves: 2 m-tile rows x 4 n-groups
__global__ __launch_bounds__(256) void k_mlp(const float* __restrict__ x,
                                             const _Float16* __restrict__ Whf,
                                             const float* __restrict__ bh,
                                             const _Float16* __restrict__ Wvf,
                                             const float* __restrict__ bv,
                                             _Float16* __restrict__ hf,
                                             _Float16* __restrict__ vf) {
  const int w = threadIdx.x >> 5, lane = threadIdx.x & 31;
  const int g = lane >> 4, ln = lane & 15;
  const _Float16* Wsel = blockIdx.y ? Wvf : Whf;
  const float*    bsel = blockIdx.y ? bv : bh;
  _Float16*       osel = blockIdx.y ? vf : hf;
  const int mt = blockIdx.x * 2 + (w >> 2);      // 16-row tile index (0..63)
  const int nb = (w & 3) * 128;                  // 128-col group
  const float* arow = x + (size_t)(mt * 16 + ln) * 1024;
  const _Float16* cbase = Wsel + (size_t)(nb + ln) * 1024;

  v8f acc[8] = {};
  for (int ks = 0; ks < 32; ++ks) {
    const int kb = ks * 32;
    v16h a = fragA_f32(arow, kb, g);
    v16h bfr[8];
#pragma unroll
    for (int j = 0; j < 8; ++j) bfr[j] = fragB_f16(cbase + j * 16 * 1024, kb, g);
#pragma unroll
    for (int j = 0; j < 8; ++j) acc[j] = WMMA_F16(a, bfr[j], acc[j]);
  }
#pragma unroll
  for (int j = 0; j < 8; ++j) {
    const int col = nb + j * 16 + ln;
    const float bias = bsel[col];
#pragma unroll
    for (int vj = 0; vj < 8; ++vj) {
      const int r = mt * 16 + vj + 8 * g;
      float val = acc[j][vj] + bias;
      val = val > 0.f ? val : 0.01f * val;       // LeakyReLU(0.01)
      osel[(size_t)r * 512 + col] = (_Float16)val;
    }
  }
}

// ============ Kernel 4: M1[bh][l][d][y] = sum_x h[b,l,h,x] * Wt[h][(d,y)][x] ============
// grid (16,16,16): x-> 1024-col chunk; y-> l-tile; z-> b*4+h. 8 waves x 128 cols each.
__global__ __launch_bounds__(256) void k_m1(const _Float16* __restrict__ hf,
                                            const _Float16* __restrict__ Wt,
                                            _Float16* __restrict__ M1) {
  const int w = threadIdx.x >> 5, lane = threadIdx.x & 31;
  const int g = lane >> 4, ln = lane & 15;
  const int bh = blockIdx.z, b = bh >> 2, h = bh & 3;
  const int lt = blockIdx.y;
  const _Float16* arow = hf + ((size_t)(b * 256 + lt * 16 + ln) * 512 + h * 128);
  const int cb = blockIdx.x * 1024 + w * 128;
  const _Float16* cbase = Wt + (size_t)h * 16384 * 128 + (size_t)(cb + ln) * 128;

  v16h a[4];
#pragma unroll
  for (int ks = 0; ks < 4; ++ks) a[ks] = fragA_f16(arow, ks * 32, g);

  v8f acc[8] = {};
#pragma unroll
  for (int ks = 0; ks < 4; ++ks) {
    const int kb = ks * 32;
    v16h bfr[8];
#pragma unroll
    for (int j = 0; j < 8; ++j) bfr[j] = fragB_f16(cbase + j * 16 * 128, kb, g);
#pragma unroll
    for (int j = 0; j < 8; ++j) acc[j] = WMMA_F16(a[ks], bfr[j], acc[j]);
  }
  _Float16* obase = M1 + (size_t)bh * 256 * 16384;
#pragma unroll
  for (int j = 0; j < 8; ++j) {
    const int col = cb + j * 16 + ln;
#pragma unroll
    for (int vj = 0; vj < 8; ++vj) {
      const int lr = lt * 16 + vj + 8 * g;
      obase[(size_t)lr * 16384 + col] = (_Float16)acc[j][vj];
    }
  }
}

// ============ Kernel 5: out[b,l,k,h*128+d] = sum_y v[b,k,h,y] * M1[bh][l][d][y] ============
// grid (1024, 4): x-> b*256+l; y-> h.  512 threads = 16 waves, wave = one 16-row k-tile x 128 d.
// M1 slice (32 KB) is shared by all 16 waves -> stage once in LDS (async copy on gfx1250).
#define SM1_PITCH 136   // 128 + 8 halves: 272B row stride -> 4-bank rotation, conflict-free b128
__global__ __launch_bounds__(512) void k_out(const _Float16* __restrict__ vf,
                                             const _Float16* __restrict__ M1,
                                             float* __restrict__ out) {
  __shared__ _Float16 sm1[128 * SM1_PITCH];
  const int w = threadIdx.x >> 5, lane = threadIdx.x & 31;
  const int g = lane >> 4, ln = lane & 15;
  const int bl = blockIdx.x, b = bl >> 8, l = bl & 255;
  const int h = blockIdx.y;
  const _Float16* m1b = M1 + (size_t)((b * 4 + h) * 256 + l) * 16384;

  // stage 128x128 f16 slice: 2048 chunks of 16B, 4 per thread
#pragma unroll
  for (int i = 0; i < 4; ++i) {
    const int c = threadIdx.x + 512 * i;
    const int row = c >> 4, seg = c & 15;
    const _Float16* gp = m1b + row * 128 + seg * 8;
    _Float16* lp = &sm1[row * SM1_PITCH + seg * 8];
#if HAVE_ASYNC_LDS
    __builtin_amdgcn_global_load_async_to_lds_b128(
        (AS_GLOBAL i4*)gp, (AS_LDS i4*)lp, 0, 0);
#else
    *(h8*)lp = *(const h8*)gp;
#endif
  }
#if HAVE_ASYNC_LDS
  ASYNC_WAIT();
#endif
  __syncthreads();

  const _Float16* arow = vf + ((size_t)(b * 256 + w * 16 + ln) * 512 + h * 128);
  v16h a[4];
#pragma unroll
  for (int ys = 0; ys < 4; ++ys) a[ys] = fragA_f16(arow, ys * 32, g);

  const _Float16* lbase = &sm1[ln * SM1_PITCH];
  v8f acc[8] = {};
#pragma unroll
  for (int ys = 0; ys < 4; ++ys) {
    const int kb = ys * 32;
    v16h bfr[8];
#pragma unroll
    for (int j = 0; j < 8; ++j) bfr[j] = fragB_16al(lbase + j * 16 * SM1_PITCH, kb, g);
#pragma unroll
    for (int j = 0; j < 8; ++j) acc[j] = WMMA_F16(a[ys], bfr[j], acc[j]);
  }

  float* obase = out + (size_t)bl * 256 * 512 + h * 128;
#pragma unroll
  for (int j = 0; j < 8; ++j) {
    const int d = j * 16 + ln;
#pragma unroll
    for (int vj = 0; vj < 8; ++vj) {
      const int k = w * 16 + vj + 8 * g;
      obase[(size_t)k * 512 + d] = acc[j][vj];   // lane = d -> coalesced f32 stores
    }
  }
}

extern "C" void kernel_launch(void* const* d_in, const int* in_sizes, int n_in,
                              void* d_out, int out_size, void* d_ws, size_t ws_size,
                              hipStream_t stream) {
  const float* x  = (const float*)d_in[0];
  const float* Wh = (const float*)d_in[1];
  const float* bh = (const float*)d_in[2];
  const float* Wv = (const float*)d_in[3];
  const float* bv = (const float*)d_in[4];
  const float* W  = (const float*)d_in[5];
  float* out = (float*)d_out;

  char* ws = (char*)d_ws;
  _Float16* Whf = (_Float16*)(ws);                          // 1 MiB
  _Float16* Wvf = (_Float16*)(ws + (1u << 20));             // 1 MiB
  _Float16* Wt  = (_Float16*)(ws + (2u << 20));             // 16 MiB
  _Float16* hf  = (_Float16*)(ws + (18u << 20));            // 1 MiB
  _Float16* vf  = (_Float16*)(ws + (19u << 20));            // 1 MiB
  _Float16* M1  = (_Float16*)(ws + (20u << 20));            // 128 MiB

  k_cvt_lin<<<4096, 256, 0, stream>>>(Wh, Wv, Whf, Wvf);
  k_cvt_W  <<<512, 256, 0, stream>>>(W, Wt);
  k_mlp    <<<dim3(32, 2), 256, 0, stream>>>(x, Whf, bh, Wvf, bv, hf, vf);
  k_m1     <<<dim3(16, 16, 16), 256, 0, stream>>>(hf, Wt, M1);
  k_out    <<<dim3(1024, 4), 512, 0, stream>>>(vf, M1, out);
}
